// RoutedExperts_18502719111701
// MI455X (gfx1250) — matrix operand order repaired and checked by reading, not compiled, not run on hardware
//
#include <hip/hip_runtime.h>

// Problem constants (match reference)
#define T_TOK  8192
#define E_EXP  8
#define K_TOP  2
#define D_IN   1024
#define H_HID  2048
#define DD_OUT 1024

typedef __attribute__((ext_vector_type(16))) __bf16 v16bf;
typedef __attribute__((ext_vector_type(8)))  float  v8f;

union Frag16 { v16bf v; unsigned u[8]; uint4 q[2]; };

__device__ __forceinline__ unsigned short f2bf(float f) {
    union { float f; unsigned u; } x; x.f = f;
    unsigned r = x.u + 0x7FFFu + ((x.u >> 16) & 1u);   // round-to-nearest-even
    return (unsigned short)(r >> 16);
}
__device__ __forceinline__ unsigned pack2(float a, float b) {
    return (unsigned)f2bf(a) | ((unsigned)f2bf(b) << 16);
}

// LDS byte offset of a shared-memory object (generic LDS pointers carry the
// DS offset in their low 32 bits on gfx12xx apertures).
__device__ __forceinline__ unsigned lds_off(const void* p) {
    return (unsigned)(unsigned long long)p;
}
// CDNA5 async copy: 16 B global -> LDS, per-lane addresses, tracked by ASYNCcnt.
__device__ __forceinline__ void async_copy16(unsigned lds_byte, const void* g) {
    asm volatile("global_load_async_to_lds_b128 %0, %1, off"
                 :: "v"(lds_byte), "v"((unsigned long long)g) : "memory");
}
__device__ __forceinline__ void wait_async0() {
    asm volatile("s_wait_asynccnt 0x0" ::: "memory");
}

// ---------------------------------------------------------------- zero output
__global__ void zero_out_kernel(float4* __restrict__ out) {
    out[(size_t)blockIdx.x * 256 + threadIdx.x] = make_float4(0.f, 0.f, 0.f, 0.f);
}

// ------------------------------------------------- fp32 -> packed bf16 convert
__global__ void cvt_bf16_kernel(const float4* __restrict__ in,
                                uint2* __restrict__ out, int n4) {
    const int i = blockIdx.x * 256 + threadIdx.x;
    if (i < n4) {
        const float4 f = in[i];
        out[i] = make_uint2(pack2(f.x, f.y), pack2(f.z, f.w));
    }
}

// ------------------------------------------------- deterministic token routing
__global__ __launch_bounds__(256) void route_kernel(
    const int* __restrict__ tki, const float* __restrict__ tkw,
    int* __restrict__ counts, int* __restrict__ tok_id_all,
    float* __restrict__ tok_w_all)
{
    const int e = blockIdx.x;
    int*   tok_id = tok_id_all + (size_t)e * T_TOK;
    float* tok_w  = tok_w_all  + (size_t)e * T_TOK;

    __shared__ int scan[256];
    __shared__ int base_s;
    const int tid = threadIdx.x;
    if (tid == 0) base_s = 0;
    __syncthreads();

    for (int t0 = 0; t0 < T_TOK; t0 += 256) {
        const int t = t0 + tid;
        float w = 0.f; int match = 0;
#pragma unroll
        for (int k = 0; k < K_TOP; ++k)
            if (tki[t * K_TOP + k] == e) { w += tkw[t * K_TOP + k]; match = 1; }
        scan[tid] = match;
        __syncthreads();
        for (int off = 1; off < 256; off <<= 1) {       // inclusive scan
            int v = (tid >= off) ? scan[tid - off] : 0;
            __syncthreads();
            scan[tid] += v;
            __syncthreads();
        }
        const int mybase = base_s;
        if (match) {
            const int pos = mybase + scan[tid] - 1;
            tok_id[pos] = t; tok_w[pos] = w;
        }
        __syncthreads();
        if (tid == 0) base_s += scan[255];
        __syncthreads();
    }
    const int c = base_s;
    if (tid == 0) counts[e] = c;
    const int padded = (c + 63) & ~63;
    for (int i = c + tid; i < padded; i += 256) { tok_id[i] = 0; tok_w[i] = 0.f; }
}

// Fragment-order tile: 4 sub-tiles (16 rows) x 32 lanes x 8 dwords = 1024 dwords.
// Writer slot for the A-operand mapping (ISA 7.12.2, 16-bit A 16x32):
//   lane = m + 16*(k0 has bit 8), words = (k0 has bit 16) ? 4..7 : 0..3
// Writer slot for the B-operand mapping (32x16, B[k][n] = W[n][k]):
//   lane = n + 16*(k0 has bit 8), words = (k0 has bit 16) ? 4..7 : 0..3
__device__ __forceinline__ unsigned slotA_dw(int r, int k0) {
    const int s = r >> 4, m = r & 15;
    return (unsigned)(s * 256 + (m + ((k0 & 8) ? 16 : 0)) * 8 + ((k0 & 16) ? 4 : 0));
}
__device__ __forceinline__ unsigned slotB_dw(int r, int k0) {
    const int s = r >> 4, n = r & 15;
    return (unsigned)(s * 256 + (n + ((k0 & 8) ? 16 : 0)) * 8 + ((k0 & 16) ? 4 : 0));
}

// ------------------------------------------------------ expert up-proj + SwiGLU
__global__ __launch_bounds__(256) void gu_kernel(
    const unsigned short* __restrict__ xb, const unsigned short* __restrict__ Wgb,
    const unsigned short* __restrict__ Wub, const int* __restrict__ count_e,
    const int* __restrict__ tok_id, unsigned short* __restrict__ h_buf)
{
    const int cnt    = count_e[0];
    const int padded = (cnt + 63) & ~63;
    const int bm = blockIdx.x * 64;
    if (bm >= padded) return;                  // uniform early-out, EXEC stays full
    const int bn = blockIdx.y * 64;

    __shared__ unsigned ldsA[2][1024];
    __shared__ unsigned ldsG[2][1024];
    __shared__ unsigned ldsU[2][1024];

    const int tid = threadIdx.x;
    const int r   = tid >> 2;                  // fill row 0..63
    const int k0  = (tid & 3) * 8;             // fill k offset

    const int tok = tok_id[bm + r];            // gathered A row (pads -> token 0)
    const unsigned short* px = xb  + (size_t)tok * D_IN + k0;
    const unsigned short* pg = Wgb + (size_t)(bn + r) * D_IN + k0;
    const unsigned short* pu = Wub + (size_t)(bn + r) * D_IN + k0;

    const unsigned slA = slotA_dw(r, k0) * 4u;         // byte offsets within tile
    const unsigned slB = slotB_dw(r, k0) * 4u;
    const unsigned offA[2] = { lds_off(&ldsA[0][0]) + slA, lds_off(&ldsA[1][0]) + slA };
    const unsigned offG[2] = { lds_off(&ldsG[0][0]) + slB, lds_off(&ldsG[1][0]) + slB };
    const unsigned offU[2] = { lds_off(&ldsU[0][0]) + slB, lds_off(&ldsU[1][0]) + slB };

    const int lane = tid & 31, wv = tid >> 5;
    const int hi = lane >> 4, ln = lane & 15;
    const int mo = (wv & 3) * 16, co = (wv >> 2) * 32;
    const int fa = (wv & 3) * 256 + lane * 8;          // A fragment dword offset
    const int fb = (wv >> 2) * 512 + lane * 8;         // first B fragment dword offset

    const v8f z8 = {0.f,0.f,0.f,0.f,0.f,0.f,0.f,0.f};
    v8f accG[2] = {z8, z8}, accU[2] = {z8, z8};

    // running next-tile pointers (steady state = pure pointer increment)
    const unsigned short* pxc = px + 32;
    const unsigned short* pgc = pg + 32;
    const unsigned short* puc = pu + 32;

    auto issue3 = [&](unsigned b) {
        async_copy16(offA[b], pxc);
        async_copy16(offG[b], pgc);
        async_copy16(offU[b], puc);
        pxc += 32; pgc += 32; puc += 32;
    };
    auto step = [&](const unsigned* tA, const unsigned* tG, const unsigned* tU) {
        Frag16 a;                               // 2x ds_load_b128
        a.q[0] = *(const uint4*)(tA + fa);
        a.q[1] = *(const uint4*)(tA + fa + 4);
#pragma unroll
        for (int c = 0; c < 2; ++c) {
            Frag16 bg, bu;
            bg.q[0] = *(const uint4*)(tG + fb + c * 256);
            bg.q[1] = *(const uint4*)(tG + fb + c * 256 + 4);
            bu.q[0] = *(const uint4*)(tU + fb + c * 256);
            bu.q[1] = *(const uint4*)(tU + fb + c * 256 + 4);
            accG[c] = __builtin_amdgcn_wmma_f32_16x16x32_bf16(
                false, a.v, false, bg.v, (short)0, accG[c], false, false);
            accU[c] = __builtin_amdgcn_wmma_f32_16x16x32_bf16(
                false, a.v, false, bu.v, (short)0, accU[c], false, false);
        }
    };

    // prologue: fill buffer 0 with tile 0
    async_copy16(offA[0], px);
    async_copy16(offG[0], pg);
    async_copy16(offU[0], pu);

    const int NK = D_IN / 32;                   // 32, even
    for (int i = 0; i < NK - 2; i += 2) {       // branch-free ping/pong steady state
        wait_async0(); __syncthreads();
        issue3(1);
        step(ldsA[0], ldsG[0], ldsU[0]);
        wait_async0(); __syncthreads();
        issue3(0);
        step(ldsA[1], ldsG[1], ldsU[1]);
    }
    wait_async0(); __syncthreads();
    issue3(1);
    step(ldsA[0], ldsG[0], ldsU[0]);
    wait_async0(); __syncthreads();
    step(ldsA[1], ldsG[1], ldsU[1]);

    // SwiGLU epilogue: stage 64x64 bf16 tile in LDS, then 16-B global stores.
    __syncthreads();
    unsigned short* stage = (unsigned short*)&ldsA[0][0];   // 8 KB, reused
#pragma unroll
    for (int c = 0; c < 2; ++c) {
        const int col = co + c * 16 + ln;
#pragma unroll
        for (int rr = 0; rr < 8; ++rr) {
            const int row = mo + rr + hi * 8;    // C/D layout: VGPR rr -> M=rr(+8)
            const float g = accG[c][rr], u = accU[c][rr];
            const float sig = __builtin_amdgcn_rcpf(1.f + __expf(-g));
            stage[row * 64 + col] = f2bf(g * sig * u);
        }
    }
    __syncthreads();
    {
        const int row = tid >> 2, c0 = (tid & 3) * 16;
        const uint4 v0 = *(const uint4*)&stage[row * 64 + c0];
        const uint4 v1 = *(const uint4*)&stage[row * 64 + c0 + 8];
        unsigned short* gp = h_buf + (size_t)(bm + row) * H_HID + bn + c0;
        *(uint4*)gp       = v0;
        *(uint4*)(gp + 8) = v1;
    }
}

// ------------------------------------------------- expert down-proj + combine
__global__ __launch_bounds__(256) void down_kernel(
    const unsigned short* __restrict__ h_buf, const unsigned short* __restrict__ Wdb,
    const int* __restrict__ count_e, const int* __restrict__ tok_id,
    const float* __restrict__ tok_w, float* __restrict__ out)
{
    const int cnt    = count_e[0];
    const int padded = (cnt + 63) & ~63;
    const int bm = blockIdx.x * 64;
    if (bm >= padded) return;
    const int bn = blockIdx.y * 64;

    __shared__ unsigned ldsH[2][1024];
    __shared__ unsigned ldsB[2][1024];

    const int tid = threadIdx.x;
    const int r  = tid >> 2;
    const int k0 = (tid & 3) * 8;

    const unsigned short* ph = h_buf + (size_t)(bm + r) * H_HID + k0;
    const unsigned short* pd = Wdb   + (size_t)(bn + r) * H_HID + k0;

    const unsigned slA = slotA_dw(r, k0) * 4u;
    const unsigned slB = slotB_dw(r, k0) * 4u;
    const unsigned offH[2] = { lds_off(&ldsH[0][0]) + slA, lds_off(&ldsH[1][0]) + slA };
    const unsigned offB[2] = { lds_off(&ldsB[0][0]) + slB, lds_off(&ldsB[1][0]) + slB };

    const int lane = tid & 31, wv = tid >> 5;
    const int hi = lane >> 4, ln = lane & 15;
    const int mo = (wv & 3) * 16, co = (wv >> 2) * 32;
    const int fa = (wv & 3) * 256 + lane * 8;
    const int fb = (wv >> 2) * 512 + lane * 8;

    const v8f z8 = {0.f,0.f,0.f,0.f,0.f,0.f,0.f,0.f};
    v8f acc[2] = {z8, z8};

    const unsigned short* phc = ph + 32;
    const unsigned short* pdc = pd + 32;

    auto issue2 = [&](unsigned b) {
        async_copy16(offH[b], phc);
        async_copy16(offB[b], pdc);
        phc += 32; pdc += 32;
    };
    auto step = [&](const unsigned* tH, const unsigned* tB) {
        Frag16 a;
        a.q[0] = *(const uint4*)(tH + fa);
        a.q[1] = *(const uint4*)(tH + fa + 4);
#pragma unroll
        for (int c = 0; c < 2; ++c) {
            Frag16 b;
            b.q[0] = *(const uint4*)(tB + fb + c * 256);
            b.q[1] = *(const uint4*)(tB + fb + c * 256 + 4);
            acc[c] = __builtin_amdgcn_wmma_f32_16x16x32_bf16(
                false, a.v, false, b.v, (short)0, acc[c], false, false);
        }
    };

    async_copy16(offH[0], ph);
    async_copy16(offB[0], pd);

    const int NK = H_HID / 32;                  // 64, even
    for (int i = 0; i < NK - 2; i += 2) {
        wait_async0(); __syncthreads();
        issue2(1);
        step(ldsH[0], ldsB[0]);
        wait_async0(); __syncthreads();
        issue2(0);
        step(ldsH[1], ldsB[1]);
    }
    wait_async0(); __syncthreads();
    issue2(1);
    step(ldsH[0], ldsB[0]);
    wait_async0(); __syncthreads();
    step(ldsH[1], ldsB[1]);

    // Scatter-add with combine weight. Experts run serialized on the stream and
    // each token appears at most once per expert list => plain RMW is race-free.
#pragma unroll
    for (int rr = 0; rr < 8; ++rr) {
        const int s = bm + mo + rr + hi * 8;
        if (s < cnt) {                          // pad slots never stored
            const int   t  = tok_id[s];
            const float w2 = tok_w[s];
            float* op = out + (size_t)t * DD_OUT + bn + co + ln;
            op[0]  += acc[0][rr] * w2;
            op[16] += acc[1][rr] * w2;
        }
    }
}

// ---------------------------------------------------------------------- launch
extern "C" void kernel_launch(void* const* d_in, const int* in_sizes, int n_in,
                              void* d_out, int out_size, void* d_ws, size_t ws_size,
                              hipStream_t stream)
{
    (void)in_sizes; (void)n_in; (void)out_size; (void)ws_size;

    const float* x   = (const float*)d_in[0];   // [T, D]
    const int*   tki = (const int*)  d_in[1];   // [T, K]
    const float* tkw = (const float*)d_in[2];   // [T, K]
    const float* Wg  = (const float*)d_in[3];   // [E, H, D]
    const float* Wu  = (const float*)d_in[4];   // [E, H, D]
    const float* Wd  = (const float*)d_in[5];   // [E, DOUT, H]
    float* out = (float*)d_out;                 // [T, DOUT]

    // Workspace layout (~145 MB):
    // counts | tok_id | tok_w | h_bf16[T][H] | x_bf16 | Wg_bf16 | Wu_bf16 | Wd_bf16
    constexpr size_t OFF_TID = 256;
    constexpr size_t OFF_TW  = OFF_TID + 4ull * E_EXP * T_TOK;
    constexpr size_t OFF_H   = OFF_TW  + 4ull * E_EXP * T_TOK;
    constexpr size_t OFF_XB  = OFF_H   + 2ull * T_TOK * H_HID;
    constexpr size_t OFF_WG  = OFF_XB  + 2ull * T_TOK * D_IN;
    constexpr size_t OFF_WU  = OFF_WG  + 2ull * E_EXP * H_HID * D_IN;
    constexpr size_t OFF_WD  = OFF_WU  + 2ull * E_EXP * H_HID * D_IN;

    char* ws = (char*)d_ws;
    int*   counts = (int*)ws;
    int*   tok_id = (int*)(ws + OFF_TID);
    float* tok_w  = (float*)(ws + OFF_TW);
    unsigned short* h_buf = (unsigned short*)(ws + OFF_H);
    unsigned short* xb    = (unsigned short*)(ws + OFF_XB);
    unsigned short* Wgb   = (unsigned short*)(ws + OFF_WG);
    unsigned short* Wub   = (unsigned short*)(ws + OFF_WU);
    unsigned short* Wdb   = (unsigned short*)(ws + OFF_WD);

    zero_out_kernel<<<(T_TOK * DD_OUT) / (4 * 256), 256, 0, stream>>>((float4*)out);

    const int n4x = T_TOK * D_IN / 4;                     // 2,097,152
    const int n4w = E_EXP * H_HID * D_IN / 4;             // 4,194,304
    cvt_bf16_kernel<<<n4x / 256, 256, 0, stream>>>((const float4*)x,  (uint2*)xb,  n4x);
    cvt_bf16_kernel<<<n4w / 256, 256, 0, stream>>>((const float4*)Wg, (uint2*)Wgb, n4w);
    cvt_bf16_kernel<<<n4w / 256, 256, 0, stream>>>((const float4*)Wu, (uint2*)Wub, n4w);
    cvt_bf16_kernel<<<n4w / 256, 256, 0, stream>>>((const float4*)Wd, (uint2*)Wdb, n4w);

    route_kernel<<<E_EXP, 256, 0, stream>>>(tki, tkw, counts, tok_id, tok_w);

    for (int e = 0; e < E_EXP; ++e) {
        gu_kernel<<<dim3(T_TOK / 64, H_HID / 64), 256, 0, stream>>>(
            xb, Wgb + (size_t)e * H_HID * D_IN, Wub + (size_t)e * H_HID * D_IN,
            counts + e, tok_id + (size_t)e * T_TOK, h_buf);
        down_kernel<<<dim3(T_TOK / 64, DD_OUT / 64), 256, 0, stream>>>(
            h_buf, Wdb + (size_t)e * DD_OUT * H_HID, counts + e,
            tok_id + (size_t)e * T_TOK, tok_w + (size_t)e * T_TOK, out);
    }
}
